// VoxelNet_64957085385363
// MI455X (gfx1250) — matrix operand order repaired and checked by reading, not compiled, not run on hardware
//
#include <hip/hip_runtime.h>
#include <stdint.h>

// ---------------------------------------------------------------------------
// VoxelNet on MI455X (gfx1250): implicit-GEMM convolutions on bf16 WMMA.
// Matrix-core bound (~117 GFLOP in the conv3ds vs ~400MB HBM @23.3TB/s).
// This revision raises compute intensity: one wave now owns ALL output-channel
// M-tiles for its 16-wide spatial tile (multi-accumulator), so each activation
// B-fragment is loaded once and reused by 4 (conv3d) / 8 (conv2d) WMMAs, and
// the D-matrix writeback is a single b128 store per (mt,kh) instead of 8 b16s.
// All BN reductions are fixed-order => deterministic. Workspace ~143.4 MB.
// ---------------------------------------------------------------------------

typedef __bf16 bf16_t;
typedef bf16_t   v16bf  __attribute__((ext_vector_type(16)));
typedef float    v8f    __attribute__((ext_vector_type(8)));
typedef uint16_t u16v8  __attribute__((ext_vector_type(8)));

// geometry
#define VD 10
#define VH 200
#define VW 176
#define DP 12
#define HP 202
#define WP 178
#define NVOX 12000
#define NPTS 35
#define HW   (VH*VW)        // 35200
#define DHW  (VD*HW)        // 352000
#define N0   (NVOX*NPTS)    // 420000 rows for BN0

#define NB0 820   // bn0 partial blocks (128 thr)
#define NB1 880   // bn1/bn2 partial blocks (256 thr); 880*4 divides 352000

// ---- bf16 helpers (bit-level, RNE) ----------------------------------------
static __device__ __forceinline__ uint16_t f2bf(float f) {
  union { float f; uint32_t u; } v; v.f = f;
  uint32_t u = v.u;
  uint32_t r = (u + 0x7FFFu + ((u >> 16) & 1u)) >> 16;
  return (uint16_t)r;
}
static __device__ __forceinline__ float bf2f(uint16_t h) {
  union { uint32_t u; float f; } v; v.u = ((uint32_t)h) << 16;
  return v.f;
}
static __device__ __forceinline__ v16bf ldfrag(const uint16_t* p) {
  return *reinterpret_cast<const v16bf*>(p);
}

// ---- workspace layout (bytes) ---------------------------------------------
static constexpr size_t VOL0_B = (size_t)DP*HP*WP*32*2;    // 27,614,208
static constexpr size_t VOL1_B = (size_t)DP*HP*WP*64*2;    // 55,228,416
static constexpr size_t XMAP_B = (size_t)HP*WP*64*2;       //  4,602,368
static constexpr size_t Y_B    = (size_t)HP*WP*128*2;      //  9,204,736
static constexpr size_t XB_B   = (size_t)DHW*64*2;         // 45,056,000
static constexpr size_t WPA_E  = 27*4*32*16;               // 55,296 elems
static constexpr size_t WPB_E  = 27*2*4*32*16;             // 110,592
static constexpr size_t WPR_E  = 9*2*8*32*16;              // 73,728

static constexpr size_t OFF_VOL0 = 0;
static constexpr size_t OFF_VOL1 = OFF_VOL0 + VOL0_B;
static constexpr size_t OFF_XMAP = OFF_VOL1 + VOL1_B;
static constexpr size_t OFF_Y    = OFF_XMAP + XMAP_B;
static constexpr size_t OFF_XB   = OFF_Y    + Y_B;         // zero region ends here
static constexpr size_t OFF_WPA  = OFF_XB   + XB_B;
static constexpr size_t OFF_WPB  = OFF_WPA  + WPA_E*2;
static constexpr size_t OFF_WPR  = OFF_WPB  + WPB_E*2;
static constexpr size_t OFF_P0   = OFF_WPR  + WPR_E*2;             // NB0*64 f32
static constexpr size_t OFF_P1   = OFF_P0 + (size_t)NB0*64*4;      // NB1*128 f32
static constexpr size_t OFF_P2   = OFF_P1 + (size_t)NB1*128*4;
static constexpr size_t OFF_SS0  = OFF_P2 + (size_t)NB1*128*4;     // 64 f32
static constexpr size_t OFF_SS1  = OFF_SS0 + 64*4;                 // 128 f32
static constexpr size_t OFF_SS2  = OFF_SS1 + 128*4;
static constexpr size_t ZERO_W   = OFF_XB / 4;   // u32 words: vol0,vol1,xmap,y

// ---------------------------------------------------------------------------
__global__ void k_zero(uint32_t* p, long n) {
  for (long i = (long)blockIdx.x*blockDim.x + threadIdx.x; i < n;
       i += (long)gridDim.x*blockDim.x) p[i] = 0u;
}

// Pack conv weights into WMMA A-fragment lane order.
// 16x32 A fragment in this code's k order: lane L holds row M=L&15; kh=L>>4;
// element j: K = kh*16 + j (contiguous; B frags use the same k order).
__global__ void k_pack(const float* wa, const float* wb, const float* wr,
                       uint16_t* pa, uint16_t* pb, uint16_t* pr) {
  int i = blockIdx.x*256 + threadIdx.x;
  if (i < (int)WPA_E) {                         // [tap27][mt4][lane32][j16]
    int j = i & 15, lane = (i >> 4) & 31, mt = (i >> 9) & 3, tap = i >> 11;
    int kh = lane >> 4, oc = mt*16 + (lane & 15);
    int K = kh*16 + j;
    pa[i] = f2bf(wa[(oc*32 + K)*27 + tap]);
  } else if (i < (int)(WPA_E + WPB_E)) {        // [tap27][ks2][mt4][lane][j]
    int k = i - (int)WPA_E;
    int j = k & 15, lane = (k >> 4) & 31, mt = (k >> 9) & 3, ks = (k >> 11) & 1,
        tap = k >> 12;
    int kh = lane >> 4, oc = mt*16 + (lane & 15);
    int K = kh*16 + j;
    pb[k] = f2bf(wb[(oc*64 + ks*32 + K)*27 + tap]);
  } else if (i < (int)(WPA_E + WPB_E + WPR_E)) {// [tap9][ks2][mt8][lane][j]
    int k = i - (int)(WPA_E + WPB_E);
    int j = k & 15, lane = (k >> 4) & 31, mt = (k >> 9) & 7, ks = (k >> 12) & 1,
        tap = k >> 13;
    int kh = lane >> 4, oc = mt*16 + (lane & 15);
    int K = kh*16 + j;
    pr[k] = f2bf(wr[(oc*64 + ks*32 + K)*9 + tap]);
  }
}

// ---- BN0: stats over the 420000 VFE-linear rows (bias cancels in BN) ------
__global__ void k_bn0_partial(const float* vox, const float* wlin, float* part) {
  __shared__ float red[128*64];
  int t = threadIdx.x;
  float sum[32], sq[32];
#pragma unroll
  for (int c = 0; c < 32; ++c) { sum[c] = 0.f; sq[c] = 0.f; }
  for (int r = blockIdx.x*128 + t; r < N0; r += NB0*128) {
    const float* p = vox + (long)r*4;
    float x0 = p[0], x1 = p[1], x2 = p[2], x3 = p[3];
#pragma unroll
    for (int c = 0; c < 32; ++c) {
      float h = x0*wlin[c*4+0] + x1*wlin[c*4+1] + x2*wlin[c*4+2] + x3*wlin[c*4+3];
      sum[c] += h; sq[c] += h*h;
    }
  }
#pragma unroll
  for (int c = 0; c < 32; ++c) { red[t*64 + c] = sum[c]; red[t*64 + 32 + c] = sq[c]; }
  __syncthreads();
  for (int s = 64; s > 0; s >>= 1) {           // fixed-order tree: deterministic
    if (t < s)
      for (int k = 0; k < 64; ++k) red[t*64 + k] += red[(t + s)*64 + k];
    __syncthreads();
  }
  if (t < 64) part[blockIdx.x*64 + t] = red[t];
}

// Finalize BN (generic): part[b*2C + c]=sum, part[b*2C + C + c]=sumsq.
// Writes fused affine: ss[c]=g*rsqrt(var+eps), ss[C+c]=beta-mean*scale.
__global__ void k_bn_final(const float* part, int nb, int C, float invN,
                           const float* g, const float* be, float* ss) {
  int c = threadIdx.x;
  if (c >= C) return;
  float s = 0.f, q = 0.f;
  for (int b = 0; b < nb; ++b) {               // serial: deterministic
    s += part[b*2*C + c];
    q += part[b*2*C + C + c];
  }
  float m  = s*invN;
  float va = q*invN - m*m;
  float sc = g[c]*rsqrtf(va + 1e-5f);
  ss[c] = sc; ss[C + c] = be[c] - m*sc;
}

// ---- VFE apply: linear -> BN -> relu -> max over 35 pts -> scatter bf16 ---
__global__ void k_vfe_scatter(const float* vox, const float* wlin,
                              const int* coords, const float* ss,
                              uint16_t* vol0) {
  int i = blockIdx.x*256 + threadIdx.x;        // exactly 12000*32 threads
  int c = i & 31, v = i >> 5;
  float w0 = wlin[c*4+0], w1 = wlin[c*4+1], w2 = wlin[c*4+2], w3 = wlin[c*4+3];
  float sc = ss[c], sh = ss[32 + c];
  float best = -3.4e38f;
  const float* p = vox + (long)v*NPTS*4;
  for (int t = 0; t < NPTS; ++t, p += 4) {
    float h = p[0]*w0 + p[1]*w1 + p[2]*w2 + p[3]*w3;
    best = fmaxf(best, fmaxf(h*sc + sh, 0.f));
  }
  int z = coords[v*3+0], y = coords[v*3+1], x = coords[v*3+2];
  vol0[((((long)(z+1)*HP) + (y+1))*WP + (x+1))*32 + c] = f2bf(best);
}

// ---- conv3d_a: 32->64, 3x3x3 implicit GEMM. One wave = 16-wide spatial tile,
// all 4 oc-tiles as accumulators: 27 B-frag loads feed 108 WMMAs. ----------
__global__ void k_conv3a(const uint16_t* vol0, const uint16_t* wpa, uint16_t* vol1) {
  int wave = blockIdx.x*8 + (threadIdx.x >> 5);  // 22,000 waves
  int lane = threadIdx.x & 31;
  int wt = wave % 11;  int tmp = wave / 11;
  int h  = tmp % VH;   int d = tmp / VH;
  int ln = lane & 15, kh = lane >> 4;
  v8f acc[4];
#pragma unroll
  for (int mt = 0; mt < 4; ++mt) acc[mt] = {};
  for (int tap = 0; tap < 27; ++tap) {
    int dz = tap / 9, r9 = tap % 9, dy = r9 / 3, dx = r9 % 3;
    v16bf B = ldfrag(vol0 + ((((long)(d+dz)*HP) + (h+dy))*WP + (wt*16 + ln + dx))*32
                          + kh*16);
#pragma unroll
    for (int mt = 0; mt < 4; ++mt) {
      v16bf A = ldfrag(wpa + ((size_t)(tap*4 + mt)*32 + lane)*16);
      acc[mt] = __builtin_amdgcn_wmma_f32_16x16x32_bf16(false, A, false, B,
                                                        (short)0, acc[mt], false, false);
    }
  }
  long base = ((((long)(d+1)*HP) + (h+1))*WP + (wt*16 + ln + 1))*64 + kh*8;
#pragma unroll
  for (int mt = 0; mt < 4; ++mt) {              // one b128 store per mt
    u16v8 o;
#pragma unroll
    for (int r = 0; r < 8; ++r) o[r] = f2bf(acc[mt][r]);
    *reinterpret_cast<u16v8*>(vol1 + base + mt*16) = o;   // raw pre-BN
  }
}

// ---- per-channel stats over 64-channel bf16 buffers (padded or linear) ----
__global__ void k_bnstat64(const uint16_t* buf, float* part, int padded) {
  __shared__ float red[512];
  int t = threadIdx.x, c = t & 63, g = t >> 6;
  float s = 0.f, q = 0.f;
  for (int p = blockIdx.x*4 + g; p < DHW; p += NB1*4) {
    long a;
    if (padded) {
      int d = p / HW, rem = p % HW, h = rem / VW, w = rem % VW;
      a = ((((long)(d+1)*HP) + (h+1))*WP + (w+1))*64 + c;
    } else {
      a = (long)p*64 + c;
    }
    float v = bf2f(buf[a]);
    s += v; q += v*v;
  }
  red[t] = s; red[256 + t] = q;
  __syncthreads();
  if (t < 64) {                                // fixed order: deterministic
    float S = red[t] + red[t+64] + red[t+128] + red[t+192];
    float Q = red[256+t] + red[256+t+64] + red[256+t+128] + red[256+t+192];
    part[blockIdx.x*128 + t] = S;
    part[blockIdx.x*128 + 64 + t] = Q;
  }
}

// ---- BN+ReLU in place on padded vol1 interior -----------------------------
__global__ void k_bnrelu_inplace(uint16_t* vol1, const float* ss) {
  long i = (long)blockIdx.x*256 + threadIdx.x;  // exactly DHW*64 threads
  int c = (int)(i & 63); long p = i >> 6;
  int d = (int)(p / HW), rem = (int)(p % HW), h = rem / VW, w = rem % VW;
  long a = ((((long)(d+1)*HP) + (h+1))*WP + (w+1))*64 + c;
  float v = bf2f(vol1[a]);
  vol1[a] = f2bf(fmaxf(v*ss[c] + ss[64 + c], 0.f));
}

// ---- conv3d_b: 64->64, 3x3x3. Per tap: 2 B frags feed 8 WMMAs. ------------
__global__ void k_conv3b(const uint16_t* vol1, const uint16_t* wpb, uint16_t* xb) {
  int wave = blockIdx.x*8 + (threadIdx.x >> 5);  // 22,000 waves
  int lane = threadIdx.x & 31;
  int wt = wave % 11;  int tmp = wave / 11;
  int h  = tmp % VH;   int d = tmp / VH;
  int ln = lane & 15, kh = lane >> 4;
  v8f acc[4];
#pragma unroll
  for (int mt = 0; mt < 4; ++mt) acc[mt] = {};
  for (int tap = 0; tap < 27; ++tap) {
    int dz = tap / 9, r9 = tap % 9, dy = r9 / 3, dx = r9 % 3;
    long brow = ((((long)(d+dz)*HP) + (h+dy))*WP + (wt*16 + ln + dx))*64 + kh*16;
    v16bf B0 = ldfrag(vol1 + brow);
    v16bf B1 = ldfrag(vol1 + brow + 32);
#pragma unroll
    for (int mt = 0; mt < 4; ++mt) {
      v16bf A0 = ldfrag(wpb + ((size_t)((tap*2 + 0)*4 + mt)*32 + lane)*16);
      acc[mt] = __builtin_amdgcn_wmma_f32_16x16x32_bf16(false, A0, false, B0,
                                                        (short)0, acc[mt], false, false);
      v16bf A1 = ldfrag(wpb + ((size_t)((tap*2 + 1)*4 + mt)*32 + lane)*16);
      acc[mt] = __builtin_amdgcn_wmma_f32_16x16x32_bf16(false, A1, false, B1,
                                                        (short)0, acc[mt], false, false);
    }
  }
  long base = (((long)d*VH + h)*VW + (wt*16 + ln))*64 + kh*8;
#pragma unroll
  for (int mt = 0; mt < 4; ++mt) {
    u16v8 o;
#pragma unroll
    for (int r = 0; r < 8; ++r) o[r] = f2bf(acc[mt][r]);
    *reinterpret_cast<u16v8*>(xb + base + mt*16) = o;     // raw pre-BN
  }
}

// ---- BN2 + ReLU + depth mean -> padded xmap -------------------------------
__global__ void k_mean_depth(const uint16_t* xb, const float* ss, uint16_t* xmap) {
  int i = blockIdx.x*256 + threadIdx.x;        // exactly HW*64 threads
  int c = i & 63, hw = i >> 6, h = hw / VW, w = hw % VW;
  float sc = ss[c], sh = ss[64 + c], a = 0.f;
  for (int d = 0; d < VD; ++d) {
    float v = bf2f(xb[((long)d*HW + hw)*64 + c]);
    a += fmaxf(v*sc + sh, 0.f);
  }
  xmap[(((long)(h+1)*WP) + (w+1))*64 + c] = f2bf(a*0.1f);
}

// ---- conv2d_r1: 64->128, 3x3 + bias + ReLU. 8 oc-tiles per wave,
// per tap: 2 B frags feed 16 WMMAs. -----------------------------------------
__global__ void k_conv_r1(const uint16_t* xmap, const uint16_t* wpr,
                          const float* b1, uint16_t* y) {
  int wave = blockIdx.x*8 + (threadIdx.x >> 5);  // 2,200 waves
  int lane = threadIdx.x & 31;
  int wt = wave % 11;  int h = wave / 11;
  int ln = lane & 15, kh = lane >> 4;
  v8f acc[8];
#pragma unroll
  for (int mt = 0; mt < 8; ++mt) acc[mt] = {};
  for (int tap = 0; tap < 9; ++tap) {
    int dy = tap / 3, dx = tap % 3;
    long brow = (((long)(h+dy)*WP) + (wt*16 + ln + dx))*64 + kh*16;
    v16bf B0 = ldfrag(xmap + brow);
    v16bf B1 = ldfrag(xmap + brow + 32);
#pragma unroll
    for (int mt = 0; mt < 8; ++mt) {
      v16bf A0 = ldfrag(wpr + ((size_t)((tap*2 + 0)*8 + mt)*32 + lane)*16);
      acc[mt] = __builtin_amdgcn_wmma_f32_16x16x32_bf16(false, A0, false, B0,
                                                        (short)0, acc[mt], false, false);
      v16bf A1 = ldfrag(wpr + ((size_t)((tap*2 + 1)*8 + mt)*32 + lane)*16);
      acc[mt] = __builtin_amdgcn_wmma_f32_16x16x32_bf16(false, A1, false, B1,
                                                        (short)0, acc[mt], false, false);
    }
  }
  long base = (((long)(h+1)*WP) + (wt*16 + ln + 1))*128 + kh*8;
#pragma unroll
  for (int mt = 0; mt < 8; ++mt) {
    u16v8 o;
#pragma unroll
    for (int r = 0; r < 8; ++r) {
      int c = mt*16 + kh*8 + r;
      o[r] = f2bf(fmaxf(acc[mt][r] + b1[c], 0.f));
    }
    *reinterpret_cast<u16v8*>(y + base + mt*16) = o;
  }
}

// ---- r2 (1x1, 128->8) + bias + box head -----------------------------------
__global__ void k_r2_post(const uint16_t* y, const float* w2, const float* b2,
                          float* out) {
  int i = blockIdx.x*256 + threadIdx.x;
  if (i >= HW) return;
  int h = i / VW, w = i % VW;
  const uint16_t* yr = y + (((long)(h+1)*WP) + (w+1))*128;
  float o[8];
#pragma unroll
  for (int c = 0; c < 8; ++c) o[c] = b2[c];
  for (int k = 0; k < 128; ++k) {
    float v = bf2f(yr[k]);
#pragma unroll
    for (int c = 0; c < 8; ++c) o[c] += v*w2[c*128 + k];
  }
  out[0*HW + i] = o[0];
  out[1*HW + i] = o[1];
  out[2*HW + i] = o[2];
  out[3*HW + i] = expf(o[3]);
  out[4*HW + i] = expf(o[4]);
  out[5*HW + i] = expf(o[5]);
  out[6*HW + i] = tanhf(o[6]);
  out[7*HW + i] = 1.f/(1.f + expf(-o[7]));
}

// ---------------------------------------------------------------------------
extern "C" void kernel_launch(void* const* d_in, const int* in_sizes, int n_in,
                              void* d_out, int out_size, void* d_ws, size_t ws_size,
                              hipStream_t stream) {
  (void)in_sizes; (void)n_in; (void)out_size; (void)ws_size;
  const float* voxels = (const float*)d_in[0];
  const int*   coords = (const int*)  d_in[1];
  const float* w_lin  = (const float*)d_in[2];
  // d_in[3] b_lin, d_in[7] b_c3a, d_in[11] b_c3b cancel inside batch-stat BN.
  const float* g_bn0  = (const float*)d_in[4];
  const float* be_bn0 = (const float*)d_in[5];
  const float* w_c3a  = (const float*)d_in[6];
  const float* g_bn1  = (const float*)d_in[8];
  const float* be_bn1 = (const float*)d_in[9];
  const float* w_c3b  = (const float*)d_in[10];
  const float* g_bn2  = (const float*)d_in[12];
  const float* be_bn2 = (const float*)d_in[13];
  const float* w_r1   = (const float*)d_in[14];
  const float* b_r1   = (const float*)d_in[15];
  const float* w_r2   = (const float*)d_in[16];
  const float* b_r2   = (const float*)d_in[17];

  char* ws = (char*)d_ws;
  uint16_t* vol0 = (uint16_t*)(ws + OFF_VOL0);
  uint16_t* vol1 = (uint16_t*)(ws + OFF_VOL1);
  uint16_t* xmap = (uint16_t*)(ws + OFF_XMAP);
  uint16_t* ybuf = (uint16_t*)(ws + OFF_Y);
  uint16_t* xb   = (uint16_t*)(ws + OFF_XB);
  uint16_t* wpa  = (uint16_t*)(ws + OFF_WPA);
  uint16_t* wpb  = (uint16_t*)(ws + OFF_WPB);
  uint16_t* wpr  = (uint16_t*)(ws + OFF_WPR);
  float* p0  = (float*)(ws + OFF_P0);
  float* p1  = (float*)(ws + OFF_P1);
  float* p2  = (float*)(ws + OFF_P2);
  float* ss0 = (float*)(ws + OFF_SS0);
  float* ss1 = (float*)(ws + OFF_SS1);
  float* ss2 = (float*)(ws + OFF_SS2);

  // 1) zero the scatter volume + all padded halos (vol0, vol1, xmap, y)
  k_zero<<<4096, 256, 0, stream>>>((uint32_t*)ws, (long)ZERO_W);
  // 2) pack weights into WMMA A-fragment layout (bf16)
  k_pack<<<(int)((WPA_E + WPB_E + WPR_E + 255)/256), 256, 0, stream>>>(
      w_c3a, w_c3b, w_r1, wpa, wpb, wpr);
  // 3) BN0 stats over 420k VFE rows (recompute linear; it's ~0.1 GFLOP)
  k_bn0_partial<<<NB0, 128, 0, stream>>>(voxels, w_lin, p0);
  k_bn_final<<<1, 64, 0, stream>>>(p0, NB0, 32, 1.f/(float)N0, g_bn0, be_bn0, ss0);
  // 4) VFE apply + max-over-points + scatter into padded bf16 volume
  k_vfe_scatter<<<NVOX*32/256, 256, 0, stream>>>(voxels, w_lin, coords, ss0, vol0);
  // 5) conv3d_a (WMMA, 4 acc/wave) -> raw vol1 interior
  k_conv3a<<<(11*VH*VD)/8, 256, 0, stream>>>(vol0, wpa, vol1);
  // 6) BN1 stats, finalize, BN+ReLU in place
  k_bnstat64<<<NB1, 256, 0, stream>>>(vol1, p1, 1);
  k_bn_final<<<1, 64, 0, stream>>>(p1, NB1, 64, 1.f/(float)DHW, g_bn1, be_bn1, ss1);
  k_bnrelu_inplace<<<(int)(((long)DHW*64)/256), 256, 0, stream>>>(vol1, ss1);
  // 7) conv3d_b (WMMA, 4 acc/wave) -> raw xb
  k_conv3b<<<(11*VH*VD)/8, 256, 0, stream>>>(vol1, wpb, xb);
  // 8) BN2 stats, finalize; BN+ReLU fused into depth-mean -> padded xmap
  k_bnstat64<<<NB1, 256, 0, stream>>>(xb, p2, 0);
  k_bn_final<<<1, 64, 0, stream>>>(p2, NB1, 64, 1.f/(float)DHW, g_bn2, be_bn2, ss2);
  k_mean_depth<<<(HW*64)/256, 256, 0, stream>>>(xb, ss2, xmap);
  // 9) conv2d 64->128 3x3 (WMMA, 8 acc/wave) + bias + ReLU -> padded y
  k_conv_r1<<<(11*VH)/8, 256, 0, stream>>>(xmap, wpr, b_r1, ybuf);
  // 10) 1x1 conv 128->8 + box head -> d_out [1,8,200,176] f32
  k_r2_post<<<(HW + 255)/256, 256, 0, stream>>>(ybuf, w_r2, b_r2, (float*)d_out);
}
